// Seq2Seq_74045236183599
// MI455X (gfx1250) — matrix operand order repaired and checked by reading, not compiled, not run on hardware
//
#include <hip/hip_runtime.h>

#define B_  32
#define S_  64
#define T_  64
#define E_  512
#define H_  1024
#define VT_ 32000
#define G3H 3072   // 3*H

typedef __attribute__((ext_vector_type(16))) __bf16 v16bf;
typedef __attribute__((ext_vector_type(8)))  float  v8f;

union FragU { v16bf v; uint4 q[2]; };

__device__ __forceinline__ unsigned short f2bf_u(float f) {
  union { float f; unsigned u; } x; x.f = f;
  unsigned r = x.u + 0x7FFFu + ((x.u >> 16) & 1u);   // round-to-nearest-even
  return (unsigned short)(r >> 16);
}
__device__ __forceinline__ float bf2f(unsigned short s) {
  union { unsigned u; float f; } x; x.u = ((unsigned)s) << 16;
  return x.f;
}

// C[M,N] (f32) = A[M,K](bf16,row-major,lda) * W[N,K](bf16,row-major,ldw)^T + bias[N]
// Each wave computes a 32x64 macro-tile: 2 M-tiles x 4 N-tiles, 8 WMMA accumulators.
struct GemmP {
  const unsigned short* A; int lda;
  const unsigned short* W; int ldw;
  float* C; int ldc;
  const float* bias;
  int M, N, K;
};

__device__ __forceinline__ void gemm_tile(const GemmP g) {
  const int lane = threadIdx.x;                 // 0..31 (wave32)
  const int ntn    = g.N >> 6;                  // macro tiles along N (64 wide)
  const int ntiles = (g.M >> 5) * ntn;          // macro tiles total (32 tall)
  const int tile   = blockIdx.x * blockDim.y + threadIdx.y;
  if (tile >= ntiles) return;                   // wave-uniform: WMMA keeps EXEC all-1
  const int mt = tile / ntn, nt = tile - mt * ntn;
  const int hh  = lane >> 4;                    // half-wave select
  const int l15 = lane & 15;

  // A 16x32 bf16 frag: lane row; halves e -> K k0 + 16*hh + e
  const unsigned short* Ap0 = g.A + (size_t)(mt * 32 + l15)      * g.lda + hh * 16;
  const unsigned short* Ap1 = g.A + (size_t)(mt * 32 + 16 + l15) * g.lda + hh * 16;
  // B 32x16 bf16 frag: lane col; halves e -> K k0 + 16*hh + e
  const unsigned short* Bp0 = g.W + (size_t)(nt * 64 + l15)      * g.ldw + hh * 16;
  const unsigned short* Bp1 = g.W + (size_t)(nt * 64 + 16 + l15) * g.ldw + hh * 16;
  const unsigned short* Bp2 = g.W + (size_t)(nt * 64 + 32 + l15) * g.ldw + hh * 16;
  const unsigned short* Bp3 = g.W + (size_t)(nt * 64 + 48 + l15) * g.ldw + hh * 16;

  v8f acc00 = {}, acc01 = {}, acc02 = {}, acc03 = {};
  v8f acc10 = {}, acc11 = {}, acc12 = {}, acc13 = {};

  for (int k0 = 0; k0 < g.K; k0 += 32) {
    FragU a0, a1, b0, b1, b2, b3;
    a0.q[0] = *(const uint4*)(Ap0 + k0);
    a0.q[1] = *(const uint4*)(Ap0 + k0 + 8);
    a1.q[0] = *(const uint4*)(Ap1 + k0);
    a1.q[1] = *(const uint4*)(Ap1 + k0 + 8);
    b0.q[0] = *(const uint4*)(Bp0 + k0);
    b0.q[1] = *(const uint4*)(Bp0 + k0 + 8);
    b1.q[0] = *(const uint4*)(Bp1 + k0);
    b1.q[1] = *(const uint4*)(Bp1 + k0 + 8);
    b2.q[0] = *(const uint4*)(Bp2 + k0);
    b2.q[1] = *(const uint4*)(Bp2 + k0 + 8);
    b3.q[0] = *(const uint4*)(Bp3 + k0);
    b3.q[1] = *(const uint4*)(Bp3 + k0 + 8);
    acc00 = __builtin_amdgcn_wmma_f32_16x16x32_bf16(false, a0.v, false, b0.v, (short)0, acc00, false, false);
    acc01 = __builtin_amdgcn_wmma_f32_16x16x32_bf16(false, a0.v, false, b1.v, (short)0, acc01, false, false);
    acc02 = __builtin_amdgcn_wmma_f32_16x16x32_bf16(false, a0.v, false, b2.v, (short)0, acc02, false, false);
    acc03 = __builtin_amdgcn_wmma_f32_16x16x32_bf16(false, a0.v, false, b3.v, (short)0, acc03, false, false);
    acc10 = __builtin_amdgcn_wmma_f32_16x16x32_bf16(false, a1.v, false, b0.v, (short)0, acc10, false, false);
    acc11 = __builtin_amdgcn_wmma_f32_16x16x32_bf16(false, a1.v, false, b1.v, (short)0, acc11, false, false);
    acc12 = __builtin_amdgcn_wmma_f32_16x16x32_bf16(false, a1.v, false, b2.v, (short)0, acc12, false, false);
    acc13 = __builtin_amdgcn_wmma_f32_16x16x32_bf16(false, a1.v, false, b3.v, (short)0, acc13, false, false);
  }

  // C layout: VGPR r -> M = r + 8*hh, N = lane&15
  v8f* accs[2][4] = {{&acc00, &acc01, &acc02, &acc03}, {&acc10, &acc11, &acc12, &acc13}};
#pragma unroll
  for (int i = 0; i < 2; ++i) {
#pragma unroll
    for (int j = 0; j < 4; ++j) {
      const int n = nt * 64 + j * 16 + l15;
      const float bv = g.bias ? g.bias[n] : 0.0f;
      float* Cp = g.C + (size_t)(mt * 32 + i * 16 + hh * 8) * g.ldc + n;
      const v8f a = *accs[i][j];
#pragma unroll
      for (int r = 0; r < 8; ++r)
        Cp[(size_t)r * g.ldc] = a[r] + bv;
    }
  }
}

__global__ void k_gemm(GemmP g) { gemm_tile(g); }

__global__ void k_dual_gemm(GemmP g0, GemmP g1) {
  gemm_tile(blockIdx.y == 0 ? g0 : g1);
}

// fp32 -> bf16 (strided source): dst[r*cols+c] = bf16(src[r*stride+c])
__global__ void k_cvt(const float* __restrict__ src, int stride,
                      unsigned short* __restrict__ dst, int cols, long long total) {
  long long idx = (long long)blockIdx.x * 256 + threadIdx.x;
  if (idx >= total) return;
  long long r = idx / cols;
  int c = (int)(idx - r * cols);
  dst[idx] = f2bf_u(src[r * stride + c]);
}

// gather src embeddings -> bf16 (B,S,E)
__global__ void k_embed_src(const int* __restrict__ src, const float* __restrict__ emb,
                            unsigned short* __restrict__ out) {
  long long idx = (long long)blockIdx.x * 256 + threadIdx.x;   // B_*S_*E_ total
  int e  = (int)(idx & (E_ - 1));
  int bs = (int)(idx >> 9);
  int tok = src[bs];
  out[idx] = f2bf_u(emb[(size_t)tok * E_ + e]);
}

__global__ void k_zero_h(float* __restrict__ h, unsigned short* __restrict__ hbf) {
  int idx = blockIdx.x * 256 + threadIdx.x;
  if (idx < B_ * H_) { h[idx] = 0.0f; hbf[idx] = 0; }
}

// fused GRU gate update: r,z,n + h'; optionally scatter h' into sequence buffers
__global__ void k_gru_gate(const float* __restrict__ gi, const float* __restrict__ gh,
                           float* __restrict__ h, unsigned short* __restrict__ hbf,
                           float* __restrict__ seq_f, int seqf_stride,
                           unsigned short* __restrict__ seq_b, int seqb_stride) {
  int idx = blockIdx.x * 256 + threadIdx.x;
  if (idx >= B_ * H_) return;
  int b = idx >> 10, j = idx & (H_ - 1);
  const float* gib = gi + (size_t)b * G3H;
  const float* ghb = gh + (size_t)b * G3H;
  float r  = 1.0f / (1.0f + __expf(-(gib[j]          + ghb[j])));
  float z  = 1.0f / (1.0f + __expf(-(gib[H_ + j]     + ghb[H_ + j])));
  float n  = tanhf(gib[2 * H_ + j] + r * ghb[2 * H_ + j]);
  float h2 = (1.0f - z) * n + z * h[idx];
  h[idx] = h2;
  unsigned short hb = f2bf_u(h2);
  hbf[idx] = hb;
  if (seq_f) seq_f[(size_t)b * seqf_stride + j] = h2;
  if (seq_b) seq_b[(size_t)b * seqb_stride + j] = hb;
}

// fused attention: energy=tanh(enc_proj + hWh), scores, softmax, ctx; builds x=[emb|ctx] bf16
__global__ void k_attn(const float* __restrict__ hWh, const float* __restrict__ enc_proj,
                       const float* __restrict__ enc_f, const float* __restrict__ vw,
                       const float* __restrict__ tgt_emb, const int* __restrict__ tgt,
                       unsigned short* __restrict__ xdec, int t) {
  const int b   = blockIdx.x;
  const int tid = threadIdx.x;        // 256 threads = 8 waves
  __shared__ float ssc[S_];
  __shared__ float sinv;
  const int lane = tid & 31, w = tid >> 5;
  const float* hwb = hWh + (size_t)b * E_;
  // scores: wave w handles rows s = 8w .. 8w+7
  for (int s = w * 8; s < w * 8 + 8; ++s) {
    const float* ep = enc_proj + ((size_t)b * S_ + s) * E_;
    float p = 0.0f;
    for (int e = lane; e < E_; e += 32)
      p += tanhf(ep[e] + hwb[e]) * vw[e];
    for (int off = 16; off > 0; off >>= 1) p += __shfl_down(p, off, 32);
    if (lane == 0) ssc[s] = p;
  }
  __syncthreads();
  if (tid == 0) {
    float mx = ssc[0];
    for (int s = 1; s < S_; ++s) mx = fmaxf(mx, ssc[s]);
    float sum = 0.0f;
    for (int s = 0; s < S_; ++s) { float e = __expf(ssc[s] - mx); ssc[s] = e; sum += e; }
    sinv = 1.0f / sum;
  }
  __syncthreads();
  const float inv = sinv;
  // ctx over H, write into x[:,512:1536]
  for (int j = tid; j < H_; j += 256) {
    const float* eb = enc_f + (size_t)b * S_ * H_ + j;
    float c = 0.0f;
#pragma unroll 4
    for (int s = 0; s < S_; ++s) c += ssc[s] * eb[(size_t)s * H_];
    xdec[(size_t)b * (E_ + H_) + E_ + j] = f2bf_u(c * inv);
  }
  // target embedding into x[:,0:512]
  const int tok = tgt[b * T_ + t];
  for (int j = tid; j < E_; j += 256)
    xdec[(size_t)b * (E_ + H_) + j] = f2bf_u(tgt_emb[(size_t)tok * E_ + j]);
}

extern "C" void kernel_launch(void* const* d_in, const int* in_sizes, int n_in,
                              void* d_out, int out_size, void* d_ws, size_t ws_size,
                              hipStream_t stream) {
  (void)in_sizes; (void)n_in; (void)out_size; (void)ws_size;
  const int*   src     = (const int*)  d_in[0];
  const int*   tgt     = (const int*)  d_in[1];
  const float* semb    = (const float*)d_in[2];
  const float* temb    = (const float*)d_in[3];
  const float* encWih  = (const float*)d_in[4];
  const float* encWhh  = (const float*)d_in[5];
  const float* enc_bih = (const float*)d_in[6];
  const float* enc_bhh = (const float*)d_in[7];
  const float* decWih  = (const float*)d_in[8];
  const float* decWhh  = (const float*)d_in[9];
  const float* dec_bih = (const float*)d_in[10];
  const float* dec_bhh = (const float*)d_in[11];
  const float* attnW   = (const float*)d_in[12];
  const float* attn_b  = (const float*)d_in[13];
  const float* v_w     = (const float*)d_in[14];
  const float* fcW     = (const float*)d_in[15];
  const float* fc_b    = (const float*)d_in[16];
  float* out = (float*)d_out;

  char* ws = (char*)d_ws;
  size_t off = 0;
  auto alloc = [&](size_t bytes) -> char* {
    char* p = ws + off;
    off = (off + bytes + 255) & ~(size_t)255;
    return p;
  };
  unsigned short* fcW_bf    = (unsigned short*)alloc((size_t)VT_ * H_ * 2);
  unsigned short* encWih_bf = (unsigned short*)alloc((size_t)G3H * E_ * 2);
  unsigned short* encWhh_bf = (unsigned short*)alloc((size_t)G3H * H_ * 2);
  unsigned short* decWih_bf = (unsigned short*)alloc((size_t)G3H * (E_ + H_) * 2);
  unsigned short* decWhh_bf = (unsigned short*)alloc((size_t)G3H * H_ * 2);
  unsigned short* Wh_bf     = (unsigned short*)alloc((size_t)E_ * H_ * 2);
  unsigned short* We_bf     = (unsigned short*)alloc((size_t)E_ * H_ * 2);
  unsigned short* xsrc_bf   = (unsigned short*)alloc((size_t)B_ * S_ * E_ * 2);
  float*          enc_f     = (float*)         alloc((size_t)B_ * S_ * H_ * 4);
  unsigned short* enc_bf    = (unsigned short*)alloc((size_t)B_ * S_ * H_ * 2);
  float*          encproj   = (float*)         alloc((size_t)B_ * S_ * E_ * 4);
  float*          h_f       = (float*)         alloc((size_t)B_ * H_ * 4);
  unsigned short* h_bf      = (unsigned short*)alloc((size_t)B_ * H_ * 2);
  float*          gi        = (float*)         alloc((size_t)B_ * G3H * 4);
  float*          gh        = (float*)         alloc((size_t)B_ * G3H * 4);
  float*          hWh       = (float*)         alloc((size_t)B_ * E_ * 4);
  unsigned short* xdec_bf   = (unsigned short*)alloc((size_t)B_ * (E_ + H_) * 2);
  unsigned short* h2all_bf  = (unsigned short*)alloc((size_t)B_ * (T_ - 1) * H_ * 2);

  auto cvt = [&](const float* s, int stride, unsigned short* d, long long rows, long long cols) {
    long long n = rows * cols;
    k_cvt<<<dim3((unsigned)((n + 255) / 256)), dim3(256), 0, stream>>>(s, stride, d, (int)cols, n);
  };
  auto gemm = [&](const unsigned short* A, int lda, const unsigned short* W, int ldw,
                  float* C, int ldc, const float* bias, int M, int N, int K) {
    GemmP g{A, lda, W, ldw, C, ldc, bias, M, N, K};
    int tiles = (M / 32) * (N / 64);
    k_gemm<<<dim3((tiles + 7) / 8), dim3(32, 8), 0, stream>>>(g);
  };
  auto dualgemm = [&](GemmP g0, GemmP g1) {
    int t0 = (g0.M / 32) * (g0.N / 64), t1 = (g1.M / 32) * (g1.N / 64);
    int mx = t0 > t1 ? t0 : t1;
    k_dual_gemm<<<dim3((mx + 7) / 8, 2), dim3(32, 8), 0, stream>>>(g0, g1);
  };

  // ---- weight conversion (fp32 -> bf16 in workspace) ----
  cvt(encWih, E_,        encWih_bf, G3H, E_);
  cvt(encWhh, H_,        encWhh_bf, G3H, H_);
  cvt(decWih, E_ + H_,   decWih_bf, G3H, E_ + H_);
  cvt(decWhh, H_,        decWhh_bf, G3H, H_);
  cvt(attnW,       2 * H_, Wh_bf, E_, H_);     // Wh = attn_W[:, :H]
  cvt(attnW + H_,  2 * H_, We_bf, E_, H_);     // We = attn_W[:, H:]
  cvt(fcW,    H_,        fcW_bf, VT_, H_);

  // ---- source embeddings + h0 ----
  k_embed_src<<<dim3((B_ * S_ * E_) / 256), dim3(256), 0, stream>>>(src, semb, xsrc_bf);
  k_zero_h<<<dim3((B_ * H_ + 255) / 256), dim3(256), 0, stream>>>(h_f, h_bf);

  // ---- encoder: 64 sequential GRU steps ----
  for (int t = 0; t < S_; ++t) {
    GemmP ggi{xsrc_bf + (size_t)t * E_, S_ * E_, encWih_bf, E_, gi, G3H, enc_bih, B_, G3H, E_};
    GemmP ggh{h_bf, H_, encWhh_bf, H_, gh, G3H, enc_bhh, B_, G3H, H_};
    dualgemm(ggi, ggh);
    k_gru_gate<<<dim3((B_ * H_) / 256), dim3(256), 0, stream>>>(
        gi, gh, h_f, h_bf, enc_f + (size_t)t * H_, S_ * H_, enc_bf + (size_t)t * H_, S_ * H_);
  }

  // ---- enc_proj = enc @ We^T + attn_b : (2048 x 512), K=1024 ----
  gemm(enc_bf, H_, We_bf, H_, encproj, E_, attn_b, B_ * S_, E_, H_);

  // ---- decoder: 63 sequential steps ----
  for (int t = 0; t < T_ - 1; ++t) {
    gemm(h_bf, H_, Wh_bf, H_, hWh, E_, nullptr, B_, E_, H_);          // h @ Wh^T
    k_attn<<<dim3(B_), dim3(256), 0, stream>>>(hWh, encproj, enc_f, v_w, temb, tgt, xdec_bf, t);
    GemmP ggi{xdec_bf, E_ + H_, decWih_bf, E_ + H_, gi, G3H, dec_bih, B_, G3H, E_ + H_};
    GemmP ggh{h_bf, H_, decWhh_bf, H_, gh, G3H, dec_bhh, B_, G3H, H_};
    dualgemm(ggi, ggh);
    k_gru_gate<<<dim3((B_ * H_) / 256), dim3(256), 0, stream>>>(
        gi, gh, h_f, h_bf, nullptr, 0, h2all_bf + (size_t)t * H_, (T_ - 1) * H_);
  }

  // ---- batched logits: (2016 x 32000), K=1024, straight into d_out ----
  gemm(h2all_bf, H_, fcW_bf, H_, out, VT_, fc_b, B_ * (T_ - 1), VT_, H_);
}
